// MultiHeadAttention_78494822301724
// MI455X (gfx1250) — matrix operand order repaired and checked by reading, not compiled, not run on hardware
//
#include <hip/hip_runtime.h>

// ---------------------------------------------------------------------------
// MHA forward for MI455X (gfx1250, wave32, WMMA).
// bf16 operands, fp32 WMMA accumulation (v_wmma_f32_16x16x32_bf16).
// GEMMs: double-buffered async-to-LDS staging of A (ASYNCcnt pipelined one
// K-step ahead), register-pipelined B, 8 WMMAs per K-step per wave.
// ---------------------------------------------------------------------------

#define T_SEQ   2048
#define C_DIM   2048
#define H_NUM   16
#define D_HEAD  128
#define B_BATCH 2
#define M_ROWS  (B_BATCH * T_SEQ)   // 4096
#define N_QKV   (3 * C_DIM)         // 6144

typedef __attribute__((ext_vector_type(16))) __bf16          v16bf;
typedef __attribute__((ext_vector_type(8)))  float           v8f;
typedef __attribute__((ext_vector_type(8)))  unsigned short  ushort8;

union V16U { v16bf v; ushort8 h[2]; };

__device__ __forceinline__ unsigned short f2bf(float f) {
    unsigned int u = __float_as_uint(f);
    u += 0x7FFFu + ((u >> 16) & 1u);          // round-to-nearest-even
    return (unsigned short)(u >> 16);
}

// A-matrix 16x32 bf16 tile from global: lane m holds row m; lanes 0-15 get
// K = k0+0..7 (v[0..7]) and k0+16..23 (v[8..15]); lanes 16-31 get +8.
__device__ __forceinline__ v16bf load_tileA(const unsigned short* __restrict__ A,
                                            int lda, int row0, int k0, int lane) {
    const int r    = lane & 15;
    const int koff = (lane >> 4) << 3;        // 0 or 8
    const ushort8* p = (const ushort8*)(A + (size_t)(row0 + r) * lda + (k0 + koff));
    V16U u; u.h[0] = p[0]; u.h[1] = p[2];     // +16 elements
    return u.v;
}

// Same A-layout read, but from an LDS-staged [64][32] bf16 tile.
__device__ __forceinline__ v16bf lds_loadA(const unsigned short* As, int i, int lane) {
    const int r    = lane & 15;
    const int koff = (lane >> 4) << 3;
    const ushort8* p = (const ushort8*)(As + (16 * i + r) * 32 + koff);
    V16U u; u.h[0] = p[0]; u.h[1] = p[2];
    return u.v;
}

// B-matrix 32x16 bf16 tile (K x N): lane n holds column n; lanes 0-15 hold
// K = k0+0..7 (v[0..7]) and k0+16..23 (v[8..15]); lanes 16-31 get +8.
// Source must be row-major with rows = N (length-K rows).
__device__ __forceinline__ v16bf load_tileB(const unsigned short* __restrict__ Bm,
                                            int ldb, int n0, int k0, int lane) {
    const int n    = lane & 15;
    const int koff = (lane >> 4) << 3;        // 0 or 8
    const ushort8* p = (const ushort8*)(Bm + (size_t)(n0 + n) * ldb + (k0 + koff));
    V16U u; u.h[0] = p[0]; u.h[1] = p[2];
    return u.v;
}

__device__ __forceinline__ v8f wmma_bf16(v16bf a, v16bf b, v8f c) {
    return __builtin_amdgcn_wmma_f32_16x16x32_bf16(false, a, false, b,
                                                   (short)0, c, false, false);
}

// CDNA5 async copy: 16 bytes per lane, global -> LDS, tracked by ASYNCcnt.
__device__ __forceinline__ void async_g2l_b128(unsigned int lds_off,
                                               const unsigned short* gptr) {
    unsigned long long ga = (unsigned long long)(uintptr_t)gptr;
    asm volatile("global_load_async_to_lds_b128 %0, %1, off"
                 :: "v"(lds_off), "v"(ga) : "memory");
}

// Reference RoPE factor: emb[d] = d<64 ? sin(t*d/64) : cos(t*(d-64)/64)
__device__ __forceinline__ float rope_emb(int t, int d) {
    if (d < 64)  return __sinf((float)t * (float)d * 0.015625f);
    return __cosf((float)t * (float)(d - 64) * 0.015625f);
}

// ---------------------------------------------------------------------------
// Kernel 0: fp32 -> bf16 conversion
// ---------------------------------------------------------------------------
__global__ __launch_bounds__(256) void f2bf_kernel(const float* __restrict__ in,
                                                   unsigned short* __restrict__ out,
                                                   int n) {
    int i = blockIdx.x * 256 + threadIdx.x;
    if (i < n) out[i] = f2bf(in[i]);
}

// ---------------------------------------------------------------------------
// GEMM core: block = 128 threads (4 waves), block tile 64(M) x 128(N).
// A k-slice [64 x 32] staged into a double-buffered LDS tile by async-to-LDS,
// shared by all 4 waves; copy for step k+1 overlaps compute on step k
// (in-order async completion => s_wait_asynccnt 2 releases the older pair).
// B register-pipelined one K-step ahead. 8 WMMAs per K-step per wave with
// 4 global b128 (B) + 8 ds b128 (A) -> 0.5 global VMEM per WMMA.
// ---------------------------------------------------------------------------
__device__ __forceinline__ void gemm_core(const unsigned short* __restrict__ Abase,
                                          const unsigned short* __restrict__ Wb,
                                          unsigned short* As,   // [2][64*32]
                                          int row0, int n0, int tid, int lane,
                                          v8f c[4][2]) {
    const int arow = tid >> 2;                 // 0..31
    const int acol = (tid & 3) << 3;           // 0,8,16,24
    const unsigned short* ag0 = Abase + (size_t)(row0 + arow) * C_DIM + acol;
    const unsigned short* ag1 = ag0 + (size_t)32 * C_DIM;
    const unsigned int l0 = (unsigned int)(uintptr_t)(As + arow * 32 + acol);
    const unsigned int l1 = l0 + 32 * 32 * 2;
    const unsigned int bufbytes = 64 * 32 * 2; // 4KB per buffer

    // prologue: stage k=0 into buffer 0, preload B(k=0)
    async_g2l_b128(l0, ag0);
    async_g2l_b128(l1, ag1);
    v16bf b0 = load_tileB(Wb, C_DIM, n0,      0, lane);
    v16bf b1 = load_tileB(Wb, C_DIM, n0 + 16, 0, lane);

    for (int k0 = 0; k0 < C_DIM; k0 += 32) {
        const int cur = (k0 >> 5) & 1;         // block-uniform
        const bool more = (k0 + 32) < C_DIM;   // block-uniform

        if (more) {                            // stage k+1 into the other buffer
            const unsigned int nxtoff = (cur ^ 1) * bufbytes;
            async_g2l_b128(l0 + nxtoff, ag0 + k0 + 32);
            async_g2l_b128(l1 + nxtoff, ag1 + k0 + 32);
            asm volatile("s_wait_asynccnt 2" ::: "memory");  // older pair done
        } else {
            asm volatile("s_wait_asynccnt 0" ::: "memory");
        }
        __syncthreads();                       // cur tile visible to all waves

        v16bf nb0 = b0, nb1 = b1;
        if (more) {                            // register-pipeline B one step ahead
            nb0 = load_tileB(Wb, C_DIM, n0,      k0 + 32, lane);
            nb1 = load_tileB(Wb, C_DIM, n0 + 16, k0 + 32, lane);
        }

        const unsigned short* Ac = As + cur * (64 * 32);
#pragma unroll
        for (int i = 0; i < 4; ++i) {
            v16bf am = lds_loadA(Ac, i, lane);
            c[i][0] = wmma_bf16(am, b0, c[i][0]);
            c[i][1] = wmma_bf16(am, b1, c[i][1]);
        }
        b0 = nb0; b1 = nb1;
        __syncthreads();   // all reads of cur done before iter+1 overwrites it
    }
}

// ---------------------------------------------------------------------------
// Kernel 1: QKV GEMM  C[4096,6144] = X @ W^T + bias, fused epilogue:
//   q -> [B,H,T,D] bf16 (RoPE * 1/sqrt(D)),  k -> [B,H,T,D] bf16 (RoPE),
//   v -> [B,H,D,T] bf16 (transposed for WMMA B-tile loads in attention)
// ---------------------------------------------------------------------------
__global__ __launch_bounds__(128) void qkv_gemm_kernel(
        const unsigned short* __restrict__ xb, const unsigned short* __restrict__ wb,
        const float* __restrict__ bias,
        unsigned short* __restrict__ qb, unsigned short* __restrict__ kb,
        unsigned short* __restrict__ vtb) {
    __shared__ unsigned short As[2 * 64 * 32];
    const int tid  = threadIdx.x;
    const int lane = tid & 31;
    const int wave = tid >> 5;
    const int n0   = blockIdx.x * 128 + wave * 32;
    const int row0 = blockIdx.y * 64;

    v8f z = {};
    v8f c[4][2] = {{z, z}, {z, z}, {z, z}, {z, z}};
    gemm_core(xb, wb, As, row0, n0, tid, lane, c);

    const int half8 = (lane >> 4) << 3;
    const int ncol  = lane & 15;
#pragma unroll
    for (int cg = 0; cg < 2; ++cg) {
        const int n    = n0 + cg * 16 + ncol;
        const int part = n >> 11;            // 0=q, 1=k, 2=v (uniform per wave)
        const int cc   = n & 2047;
        const int h    = cc >> 7;
        const int d    = cc & 127;
        const float bv = bias[n];
#pragma unroll
        for (int i = 0; i < 4; ++i) {
#pragma unroll
            for (int j = 0; j < 8; ++j) {
                const int m    = row0 + 16 * i + j + half8;
                const int bidx = m >> 11;
                const int t    = m & (T_SEQ - 1);
                float val = c[i][cg][j] + bv;
                const size_t hd = (size_t)(bidx * H_NUM + h);
                if (part == 0) {
                    val *= rope_emb(t, d) * 0.08838834764831845f;   // 1/sqrt(128)
                    qb[(hd * T_SEQ + t) * D_HEAD + d] = f2bf(val);
                } else if (part == 1) {
                    val *= rope_emb(t, d);
                    kb[(hd * T_SEQ + t) * D_HEAD + d] = f2bf(val);
                } else {
                    vtb[(hd * D_HEAD + d) * T_SEQ + t] = f2bf(val);
                }
            }
        }
    }
}

// ---------------------------------------------------------------------------
// Kernel 2: causal flash attention. Block = 4 waves, each wave owns a 16-query
// tile and streams keys in 32-wide steps with online softmax. 16 WMMAs per step
// (8 QK^T + 8 PV). P transposed C-layout -> A-layout through wave-private LDS.
// ---------------------------------------------------------------------------
__global__ __launch_bounds__(128) void attn_kernel(
        const unsigned short* __restrict__ qb, const unsigned short* __restrict__ kb,
        const unsigned short* __restrict__ vtb, unsigned short* __restrict__ attnb) {
    __shared__ unsigned short lds_p[4][16 * 32];

    const int lane = threadIdx.x & 31;
    const int wave = threadIdx.x >> 5;
    const int bh   = blockIdx.y;              // b*16 + h
    const int bidx = bh >> 4;
    const int h    = bh & 15;
    const int m0   = blockIdx.x * 64 + wave * 16;

    const unsigned short* qh = qb  + (size_t)bh * T_SEQ * D_HEAD;   // [T,D]
    const unsigned short* kh = kb  + (size_t)bh * T_SEQ * D_HEAD;   // [T,D]
    const unsigned short* vh = vtb + (size_t)bh * D_HEAD * T_SEQ;   // [D,T]

    v16bf qa[4];
#pragma unroll
    for (int kc = 0; kc < 4; ++kc) qa[kc] = load_tileA(qh, D_HEAD, m0, kc * 32, lane);

    v8f z = {};
    v8f o[8] = {z, z, z, z, z, z, z, z};      // 16 x 128 output accumulator
    float mrow[8], lrow[8];
#pragma unroll
    for (int j = 0; j < 8; ++j) { mrow[j] = -3.0e38f; lrow[j] = 0.0f; }

    const int half8 = (lane >> 4) << 3;
    const int ncol  = lane & 15;
    unsigned short* Lp = lds_p[wave];

    for (int n0 = 0; n0 <= m0; n0 += 32) {
        // ---- scores: two 16-key tiles, K = D = 128 (4 x K32 WMMA each) ----
        v8f c0 = z, c1 = z;
#pragma unroll
        for (int kc = 0; kc < 4; ++kc)
            c0 = wmma_bf16(qa[kc], load_tileB(kh, D_HEAD, n0,      kc * 32, lane), c0);
#pragma unroll
        for (int kc = 0; kc < 4; ++kc)
            c1 = wmma_bf16(qa[kc], load_tileB(kh, D_HEAD, n0 + 16, kc * 32, lane), c1);

        // ---- causal mask + online softmax (row reductions within 16-lane half)
#pragma unroll
        for (int j = 0; j < 8; ++j) {
            const int m = m0 + j + half8;
            float s0 = c0[j]; if (n0 + ncol      > m) s0 = -3.0e38f;
            float s1 = c1[j]; if (n0 + 16 + ncol > m) s1 = -3.0e38f;
            float mx = fmaxf(s0, s1);
#pragma unroll
            for (int off = 1; off < 16; off <<= 1) mx = fmaxf(mx, __shfl_xor(mx, off, 32));
            const float mnew  = fmaxf(mrow[j], mx);
            const float scale = __expf(mrow[j] - mnew);
            const float p0 = __expf(s0 - mnew);
            const float p1 = __expf(s1 - mnew);
            float rs = p0 + p1;
#pragma unroll
            for (int off = 1; off < 16; off <<= 1) rs += __shfl_xor(rs, off, 32);
            lrow[j] = lrow[j] * scale + rs;
            mrow[j] = mnew;
#pragma unroll
            for (int t = 0; t < 8; ++t) o[t][j] *= scale;
            // stash P (bf16) in wave-private LDS, C-layout addressing
            Lp[(j + half8) * 32 + ncol]      = f2bf(p0);
            Lp[(j + half8) * 32 + 16 + ncol] = f2bf(p1);
        }

        // DS ops are in-order per wave; fence compiler + wait stores before reads
        __builtin_amdgcn_wave_barrier();
        asm volatile("s_wait_dscnt 0" ::: "memory");

        // ---- reload P in A-matrix layout (16 x 32) ----
        v16bf pa;
        {
            const ushort8* p8 = (const ushort8*)(Lp + (lane & 15) * 32 + half8);
            V16U u; u.h[0] = p8[0]; u.h[1] = p8[2];
            pa = u.v;
        }
        __builtin_amdgcn_wave_barrier();

        // ---- PV: O[16,128] += P[16,32] @ V[32,128], 8 d-tiles ----
#pragma unroll
        for (int t = 0; t < 8; ++t) {
            v16bf bv = load_tileB(vh, T_SEQ, t * 16, n0, lane);   // B[k][n]=vT[d][key]
            o[t] = wmma_bf16(pa, bv, o[t]);
        }
    }

    // ---- normalize and store as bf16 [B*T, C] (row = b*T + t, col = h*128 + d)
#pragma unroll
    for (int j = 0; j < 8; ++j) {
        const float invl = 1.0f / lrow[j];
        const size_t row = (size_t)bidx * T_SEQ + (m0 + j + half8);
#pragma unroll
        for (int t = 0; t < 8; ++t) {
            const int col = h * D_HEAD + t * 16 + ncol;
            attnb[row * C_DIM + col] = f2bf(o[t][j] * invl);
        }
    }
}

// ---------------------------------------------------------------------------
// Kernel 3: output projection  out[4096,2048] = attn[4096,2048] @ W^T + bias
// ---------------------------------------------------------------------------
__global__ __launch_bounds__(128) void out_gemm_kernel(
        const unsigned short* __restrict__ ab, const unsigned short* __restrict__ wb,
        const float* __restrict__ bias, float* __restrict__ out) {
    __shared__ unsigned short As[2 * 64 * 32];
    const int tid  = threadIdx.x;
    const int lane = tid & 31;
    const int wave = tid >> 5;
    const int n0   = blockIdx.x * 128 + wave * 32;
    const int row0 = blockIdx.y * 64;

    v8f z = {};
    v8f c[4][2] = {{z, z}, {z, z}, {z, z}, {z, z}};
    gemm_core(ab, wb, As, row0, n0, tid, lane, c);

    const int half8 = (lane >> 4) << 3;
    const int ncol  = lane & 15;
#pragma unroll
    for (int cg = 0; cg < 2; ++cg) {
        const int n    = n0 + cg * 16 + ncol;
        const float bv = bias[n];
#pragma unroll
        for (int i = 0; i < 4; ++i)
#pragma unroll
            for (int j = 0; j < 8; ++j) {
                const int m = row0 + 16 * i + j + half8;
                out[(size_t)m * C_DIM + n] = c[i][cg][j] + bv;
            }
    }
}

// ---------------------------------------------------------------------------
extern "C" void kernel_launch(void* const* d_in, const int* in_sizes, int n_in,
                              void* d_out, int out_size, void* d_ws, size_t ws_size,
                              hipStream_t stream) {
    const float* x      = (const float*)d_in[0];   // [B,T,C]
    const float* qkv_w  = (const float*)d_in[1];   // [3C,C]
    const float* qkv_b  = (const float*)d_in[2];   // [3C]
    const float* out_w  = (const float*)d_in[3];   // [C,C]
    const float* out_b  = (const float*)d_in[4];   // [C]
    float* out = (float*)d_out;

    // Workspace layout (bf16 elements), total ~112 MiB
    unsigned short* ws16  = (unsigned short*)d_ws;
    unsigned short* xb    = ws16;                                   // 4096*2048
    unsigned short* wqkvb = xb    + (size_t)M_ROWS * C_DIM;         // 6144*2048
    unsigned short* wob   = wqkvb + (size_t)N_QKV  * C_DIM;         // 2048*2048
    unsigned short* qb    = wob   + (size_t)C_DIM  * C_DIM;         // B*H*T*D
    unsigned short* kb    = qb    + (size_t)M_ROWS * C_DIM;
    unsigned short* vtb   = kb    + (size_t)M_ROWS * C_DIM;
    unsigned short* attnb = vtb   + (size_t)M_ROWS * C_DIM;

    const int NX  = M_ROWS * C_DIM;      // 8,388,608
    const int NWQ = N_QKV  * C_DIM;      // 12,582,912
    const int NWO = C_DIM  * C_DIM;      // 4,194,304

    f2bf_kernel<<<(NX  + 255) / 256, 256, 0, stream>>>(x,     xb,    NX);
    f2bf_kernel<<<(NWQ + 255) / 256, 256, 0, stream>>>(qkv_w, wqkvb, NWQ);
    f2bf_kernel<<<(NWO + 255) / 256, 256, 0, stream>>>(out_w, wob,   NWO);

    qkv_gemm_kernel<<<dim3(N_QKV / 128, M_ROWS / 64), 128, 0, stream>>>(
        xb, wqkvb, qkv_b, qb, kb, vtb);

    attn_kernel<<<dim3(T_SEQ / 64, B_BATCH * H_NUM), 128, 0, stream>>>(
        qb, kb, vtb, attnb);

    out_gemm_kernel<<<dim3(C_DIM / 128, M_ROWS / 64), 128, 0, stream>>>(
        attnb, wob, out_b, out);
}
